// EnhancedSelfAttention_78451872629406
// MI455X (gfx1250) — matrix operand order repaired and checked by reading, not compiled, not run on hardware
//
#include <hip/hip_runtime.h>

typedef __attribute__((ext_vector_type(16))) __bf16 bf16x16;
typedef __attribute__((ext_vector_type(8)))  __bf16 bf16x8;
typedef __attribute__((ext_vector_type(8)))  float  f32x8;

#define WMMA_BF16(A, B, C) \
  __builtin_amdgcn_wmma_f32_16x16x32_bf16(false, (A), false, (B), (short)0, (C), false, false)

// Load a 16-element bf16 fragment as two 16-byte (b128) chunks.
static __device__ __forceinline__ bf16x16 ldfrag(const __bf16* p0, const __bf16* p1) {
  bf16x8 lo = *(const bf16x8*)p0;
  bf16x8 hi = *(const bf16x8*)p1;
  return __builtin_shufflevector(lo, hi, 0,1,2,3,4,5,6,7,8,9,10,11,12,13,14,15);
}

// ds_swizzle XOR lane shuffle (group-of-32 mode: and=0x1f, or=0, xor=M<<10).
template <int IMM>
static __device__ __forceinline__ float swz(float v) {
  int i = __builtin_bit_cast(int, v);
  i = __builtin_amdgcn_ds_swizzle(i, IMM);
  return __builtin_bit_cast(float, i);
}
// Reductions across each 16-lane half (xor 1,2,4,8 never crosses bit4).
static __device__ __forceinline__ float redmax16(float v) {
  v = fmaxf(v, swz<0x041F>(v)); v = fmaxf(v, swz<0x081F>(v));
  v = fmaxf(v, swz<0x101F>(v)); v = fmaxf(v, swz<0x201F>(v));
  return v;
}
static __device__ __forceinline__ float redsum16(float v) {
  v += swz<0x041F>(v); v += swz<0x081F>(v);
  v += swz<0x101F>(v); v += swz<0x201F>(v);
  return v;
}

// ---------------------------------------------------------------- casts ----
__global__ __launch_bounds__(256) void cast_bf16_kernel(const float* __restrict__ src,
                                                        __bf16* __restrict__ dst, int n) {
  int i = blockIdx.x * 256 + threadIdx.x;
  if (i < n) dst[i] = (__bf16)src[i];
}

// Tiled transpose+cast: src [R][C] fp32 -> dst [C][R] bf16. Coalesced both sides.
__global__ __launch_bounds__(256) void transpose_cast_kernel(const float* __restrict__ src,
                                                             __bf16* __restrict__ dst,
                                                             int R, int C) {
  __shared__ float tile[32][33];
  int tx = threadIdx.x & 31;
  int ty = threadIdx.x >> 5;                     // 0..7
  int c0 = blockIdx.x * 32;
  int r0 = blockIdx.y * 32;
#pragma unroll
  for (int i = 0; i < 32; i += 8)
    tile[ty + i][tx] = src[(size_t)(r0 + ty + i) * C + (c0 + tx)];
  __syncthreads();
#pragma unroll
  for (int i = 0; i < 32; i += 8)
    dst[(size_t)(c0 + ty + i) * R + (r0 + tx)] = (__bf16)tile[tx][ty + i];
}

// Tiled bf16 transpose per head-slice: src [BH][2048][64] -> dst [BH][64][2048].
__global__ __launch_bounds__(256) void transpose_v_kernel(const __bf16* __restrict__ src,
                                                          __bf16* __restrict__ dst) {
  __shared__ __bf16 tile[32][34];
  int tx = threadIdx.x & 31;
  int ty = threadIdx.x >> 5;
  int d0 = blockIdx.x * 32;                      // 0 or 32
  int t0 = blockIdx.y * 32;                      // 64 blocks
  const __bf16* s = src + (size_t)blockIdx.z * 2048 * 64;
  __bf16* d = dst + (size_t)blockIdx.z * 64 * 2048;
#pragma unroll
  for (int i = 0; i < 32; i += 8)
    tile[ty + i][tx] = s[(size_t)(t0 + ty + i) * 64 + (d0 + tx)];
  __syncthreads();
#pragma unroll
  for (int i = 0; i < 32; i += 8)
    d[(size_t)(d0 + ty + i) * 2048 + (t0 + tx)] = tile[tx][ty + i];
}

// ------------------------------------------- wave GEMM 32x64, double-buffered
// A: [M][K] bf16 row-major.  BT: [N][K] bf16 (transposed weights).
static __device__ __forceinline__ void gemm32x64(const __bf16* __restrict__ A,
                                                 const __bf16* __restrict__ BT,
                                                 int K, int mbase, int nbase,
                                                 f32x8 acc[8]) {
  int lane = threadIdx.x & 31;
  int hl = lane >> 4, l16 = lane & 15;
  int ha = 16 * hl;
  const __bf16* a0 = A + (size_t)(mbase +      l16) * K;
  const __bf16* a1 = A + (size_t)(mbase + 16 + l16) * K;
  const __bf16* b0 = BT + (size_t)(nbase +      l16) * K + 16 * hl;
  const __bf16* b1 = BT + (size_t)(nbase + 16 + l16) * K + 16 * hl;
  const __bf16* b2 = BT + (size_t)(nbase + 32 + l16) * K + 16 * hl;
  const __bf16* b3 = BT + (size_t)(nbase + 48 + l16) * K + 16 * hl;
  // preload k = 0
  bf16x16 ca0 = ldfrag(a0 + ha, a0 + 8 + ha);
  bf16x16 ca1 = ldfrag(a1 + ha, a1 + 8 + ha);
  bf16x16 cb0 = ldfrag(b0, b0 + 8);
  bf16x16 cb1 = ldfrag(b1, b1 + 8);
  bf16x16 cb2 = ldfrag(b2, b2 + 8);
  bf16x16 cb3 = ldfrag(b3, b3 + 8);
  for (int k0 = 32; k0 < K; k0 += 32) {
    // issue next iteration's loads before consuming current fragments
    bf16x16 na0 = ldfrag(a0 + k0 + ha, a0 + k0 + 8 + ha);
    bf16x16 na1 = ldfrag(a1 + k0 + ha, a1 + k0 + 8 + ha);
    bf16x16 nb0 = ldfrag(b0 + k0, b0 + k0 + 8);
    bf16x16 nb1 = ldfrag(b1 + k0, b1 + k0 + 8);
    bf16x16 nb2 = ldfrag(b2 + k0, b2 + k0 + 8);
    bf16x16 nb3 = ldfrag(b3 + k0, b3 + k0 + 8);
    acc[0] = WMMA_BF16(ca0, cb0, acc[0]);
    acc[1] = WMMA_BF16(ca0, cb1, acc[1]);
    acc[2] = WMMA_BF16(ca0, cb2, acc[2]);
    acc[3] = WMMA_BF16(ca0, cb3, acc[3]);
    acc[4] = WMMA_BF16(ca1, cb0, acc[4]);
    acc[5] = WMMA_BF16(ca1, cb1, acc[5]);
    acc[6] = WMMA_BF16(ca1, cb2, acc[6]);
    acc[7] = WMMA_BF16(ca1, cb3, acc[7]);
    ca0 = na0; ca1 = na1;
    cb0 = nb0; cb1 = nb1; cb2 = nb2; cb3 = nb3;
  }
  acc[0] = WMMA_BF16(ca0, cb0, acc[0]);
  acc[1] = WMMA_BF16(ca0, cb1, acc[1]);
  acc[2] = WMMA_BF16(ca0, cb2, acc[2]);
  acc[3] = WMMA_BF16(ca0, cb3, acc[3]);
  acc[4] = WMMA_BF16(ca1, cb0, acc[4]);
  acc[5] = WMMA_BF16(ca1, cb1, acc[5]);
  acc[6] = WMMA_BF16(ca1, cb2, acc[6]);
  acc[7] = WMMA_BF16(ca1, cb3, acc[7]);
}

// ------------------------------------------------------------- QKV GEMM ----
// qkv = xb[4096][1024] @ w_qkvT -> Q, K, Vtmp all as [B*H][2048][64] (coalesced).
__global__ __launch_bounds__(256) void qkv_gemm_kernel(const __bf16* __restrict__ xb,
                                                       const __bf16* __restrict__ wT,
                                                       __bf16* __restrict__ q,
                                                       __bf16* __restrict__ k,
                                                       __bf16* __restrict__ vtmp) {
  int w = (blockIdx.x * 256 + threadIdx.x) >> 5;   // 6144 waves
  int mbase = (w / 48) * 32;
  int nbase = (w % 48) * 64;
  f32x8 acc[8] = {};
  gemm32x64(xb, wT, 1024, mbase, nbase, acc);

  int lane = threadIdx.x & 31, hl = lane >> 4, l16 = lane & 15;
  int sec = nbase >> 10;                           // 0=q 1=k 2=v (uniform)
  __bf16* dst = (sec == 0) ? q : ((sec == 1) ? k : vtmp);
#pragma unroll
  for (int mt = 0; mt < 2; mt++) {
#pragma unroll
    for (int t = 0; t < 4; t++) {
      int n = nbase + t * 16 + l16;
      int c = n & 1023, h = c >> 6, d = c & 63;
#pragma unroll
      for (int r = 0; r < 8; r++) {
        int m = mbase + mt * 16 + 8 * hl + r;
        int b = m >> 11, tt = m & 2047;
        dst[(((size_t)(b * 16 + h)) * 2048 + tt) * 64 + d] = (__bf16)acc[mt * 4 + t][r];
      }
    }
  }
}

// ------------------------------------------------- flash attention (ALiBi) -
// One wave per (b,h,qtile). Q,K:[BH][2048][64], V:[BH][64][2048].
__global__ __launch_bounds__(32) void attn_kernel(const __bf16* __restrict__ q,
                                                  const __bf16* __restrict__ k,
                                                  const __bf16* __restrict__ v,
                                                  __bf16* __restrict__ ao) {
  __shared__ __bf16 pbuf[16][40];                  // P tile, padded rows (80B)
  int lane = threadIdx.x;
  int hl = lane >> 4, l16 = lane & 15;
  int idx = blockIdx.x;                            // 4096 = 32 bh * 128 qtiles
  int qtile = idx & 127;
  int bh = idx >> 7;
  int h = bh & 15;
  const float scale = 0.125f;                      // 1/sqrt(64)
  const float slope = exp2f(-0.5f * (float)(h + 1));

  // Hoisted Q A-fragments (d=0..31 and d=32..63)
  const __bf16* qrow = q + ((size_t)bh * 2048 + qtile * 16 + l16) * 64;
  bf16x16 qa0 = ldfrag(qrow + 16 * hl,      qrow + 8 + 16 * hl);
  bf16x16 qa1 = ldfrag(qrow + 32 + 16 * hl, qrow + 40 + 16 * hl);

  f32x8 oacc[4] = {};
  float mrow[8], lrow[8];
#pragma unroll
  for (int r = 0; r < 8; r++) { mrow[r] = -1e30f; lrow[r] = 0.0f; }

  const __bf16* kbh = k + (size_t)bh * 2048 * 64;
  const __bf16* vbh = v + (size_t)bh * 64 * 2048;
  int nkt = (qtile >> 1) + 1;                      // causal: 32-key tiles

  // preload K fragments for kt = 0
  const __bf16* kr0 = kbh + (size_t)(l16) * 64 + 16 * hl;
  const __bf16* kr1 = kbh + (size_t)(16 + l16) * 64 + 16 * hl;
  bf16x16 kf00 = ldfrag(kr0,      kr0 + 8);        // keys 0-15, d 0-31
  bf16x16 kf01 = ldfrag(kr0 + 32, kr0 + 40);       // keys 0-15, d 32-63
  bf16x16 kf10 = ldfrag(kr1,      kr1 + 8);        // keys 16-31, d 0-31
  bf16x16 kf11 = ldfrag(kr1 + 32, kr1 + 40);       // keys 16-31, d 32-63

  for (int kt = 0; kt < nkt; ++kt) {
    int kbase = kt * 32;
    // S = Q @ K^T : two 16x16 key-tiles, K-dim = 64 in 2 steps each
    f32x8 s0 = {}, s1 = {};
    s0 = WMMA_BF16(qa0, kf00, s0);
    s0 = WMMA_BF16(qa1, kf01, s0);
    s1 = WMMA_BF16(qa0, kf10, s1);
    s1 = WMMA_BF16(qa1, kf11, s1);

    // Issue next-tile K loads (clamped, branch-free) and this-tile V loads now;
    // their latency hides under the softmax VALU/LDS work below.
    int knb = (kt + 1 < nkt) ? (kbase + 32) : kbase;
    const __bf16* nr0 = kbh + (size_t)(knb + l16) * 64 + 16 * hl;
    const __bf16* nr1 = kbh + (size_t)(knb + 16 + l16) * 64 + 16 * hl;
    bf16x16 nf00 = ldfrag(nr0,      nr0 + 8);
    bf16x16 nf01 = ldfrag(nr0 + 32, nr0 + 40);
    bf16x16 nf10 = ldfrag(nr1,      nr1 + 8);
    bf16x16 nf11 = ldfrag(nr1 + 32, nr1 + 40);
    const __bf16* vr0 = vbh + (size_t)(     l16) * 2048 + kbase + 16 * hl;
    const __bf16* vr1 = vbh + (size_t)(16 + l16) * 2048 + kbase + 16 * hl;
    const __bf16* vr2 = vbh + (size_t)(32 + l16) * 2048 + kbase + 16 * hl;
    const __bf16* vr3 = vbh + (size_t)(48 + l16) * 2048 + kbase + 16 * hl;
    bf16x16 vf0 = ldfrag(vr0, vr0 + 8);
    bf16x16 vf1 = ldfrag(vr1, vr1 + 8);
    bf16x16 vf2 = ldfrag(vr2, vr2 + 8);
    bf16x16 vf3 = ldfrag(vr3, vr3 + 8);

    // online softmax, row-wise (row m = r + 8*hl lives in one 16-lane half)
    float kj0 = (float)(kbase + l16);
    float kj1 = kj0 + 16.0f;
#pragma unroll
    for (int r = 0; r < 8; r++) {
      float qi = (float)(qtile * 16 + 8 * hl + r);
      float v0 = s0[r] * scale + slope * fabsf(qi - kj0);
      float v1 = s1[r] * scale + slope * fabsf(qi - kj1);
      if (kj0 > qi) v0 = -1e30f;
      if (kj1 > qi) v1 = -1e30f;
      float mx = redmax16(fmaxf(v0, v1));
      float mnew = fmaxf(mrow[r], mx);
      float corr = __expf(mrow[r] - mnew);
      float p0 = __expf(v0 - mnew);
      float p1 = __expf(v1 - mnew);
      float ps = redsum16(p0 + p1);
      lrow[r] = lrow[r] * corr + ps;
      mrow[r] = mnew;
      oacc[0][r] *= corr; oacc[1][r] *= corr;
      oacc[2][r] *= corr; oacc[3][r] *= corr;
      pbuf[8 * hl + r][l16]      = (__bf16)p0;
      pbuf[8 * hl + r][16 + l16] = (__bf16)p1;
    }
    __syncthreads();                               // single-wave WG: LDS fence
    // P (16x32) as A-fragment from LDS, O += P @ V (4 d-tiles)
    const __bf16* prow = &pbuf[l16][0];
    bf16x16 pa = ldfrag(prow + 16 * hl, prow + 8 + 16 * hl);
    oacc[0] = WMMA_BF16(pa, vf0, oacc[0]);
    oacc[1] = WMMA_BF16(pa, vf1, oacc[1]);
    oacc[2] = WMMA_BF16(pa, vf2, oacc[2]);
    oacc[3] = WMMA_BF16(pa, vf3, oacc[3]);
    __syncthreads();
    kf00 = nf00; kf01 = nf01; kf10 = nf10; kf11 = nf11;
  }
  // epilogue: normalize, merge heads into [B][T][C] bf16
  int bq = bh >> 4;
#pragma unroll
  for (int dt = 0; dt < 4; dt++) {
#pragma unroll
    for (int r = 0; r < 8; r++) {
      float val = oacc[dt][r] / lrow[r];
      int tq = qtile * 16 + 8 * hl + r;
      int c = h * 64 + dt * 16 + l16;
      ao[((size_t)bq * 2048 + tq) * 1024 + c] = (__bf16)val;
    }
  }
}

// ------------------------------------------------------- output projection -
__global__ __launch_bounds__(256) void proj_gemm_kernel(const __bf16* __restrict__ ao,
                                                        const __bf16* __restrict__ wT,
                                                        float* __restrict__ out) {
  int w = (blockIdx.x * 256 + threadIdx.x) >> 5;   // 2048 waves
  int mbase = (w >> 4) * 32;
  int nbase = (w & 15) * 64;
  f32x8 acc[8] = {};
  gemm32x64(ao, wT, 1024, mbase, nbase, acc);
  int lane = threadIdx.x & 31, hl = lane >> 4, l16 = lane & 15;
#pragma unroll
  for (int mt = 0; mt < 2; mt++) {
#pragma unroll
    for (int t = 0; t < 4; t++) {
      int n = nbase + t * 16 + l16;
#pragma unroll
      for (int r = 0; r < 8; r++) {
        int m = mbase + mt * 16 + 8 * hl + r;
        out[(size_t)m * 1024 + n] = acc[mt * 4 + t][r];
      }
    }
  }
}

// ---------------------------------------------------------------- driver ---
extern "C" void kernel_launch(void* const* d_in, const int* in_sizes, int n_in,
                              void* d_out, int out_size, void* d_ws, size_t ws_size,
                              hipStream_t stream) {
  (void)in_sizes; (void)n_in; (void)out_size; (void)ws_size;
  const float* x      = (const float*)d_in[0];   // [2,2048,1024]
  const float* w_qkv  = (const float*)d_in[1];   // [1024,3072]
  const float* w_proj = (const float*)d_in[2];   // [1024,1024]

  char* ws = (char*)d_ws;
  __bf16* xb     = (__bf16*)(ws);                          //  8 MB [4096][1024]
  __bf16* wqkvT  = (__bf16*)(ws + 8388608);                //  6 MB [3072][1024]
  __bf16* wprojT = (__bf16*)(ws + 14680064);               //  2 MB [1024][1024]
  __bf16* qb     = (__bf16*)(ws + 16777216);               //  8 MB [32][2048][64]
  __bf16* kb     = (__bf16*)(ws + 25165824);               //  8 MB [32][2048][64]
  __bf16* vtmp   = (__bf16*)(ws + 33554432);               //  8 MB [32][2048][64]
  __bf16* vb     = (__bf16*)(ws + 41943040);               //  8 MB [32][64][2048]
  __bf16* aob    = (__bf16*)(ws + 50331648);               //  8 MB [4096][1024]

  cast_bf16_kernel<<<4194304 / 256, 256, 0, stream>>>(x, xb, 4194304);
  transpose_cast_kernel<<<dim3(96, 32), 256, 0, stream>>>(w_qkv, wqkvT, 1024, 3072);
  transpose_cast_kernel<<<dim3(32, 32), 256, 0, stream>>>(w_proj, wprojT, 1024, 1024);

  qkv_gemm_kernel<<<768, 256, 0, stream>>>(xb, wqkvT, qb, kb, vtmp);
  transpose_v_kernel<<<dim3(2, 64, 32), 256, 0, stream>>>(vtmp, vb);
  attn_kernel<<<4096, 32, 0, stream>>>(qb, kb, vb, aob);
  proj_gemm_kernel<<<256, 256, 0, stream>>>(aob, wprojT, (float*)d_out);
}
